// GraphEncoder_40450001994090
// MI455X (gfx1250) — compile-verified, matrix operand order checked by reading, not run on hardware
//
#include <hip/hip_runtime.h>

// Problem constants (from reference setup_inputs)
#define O_  5000     // objects
#define T_  20000    // triples
#define D_  128
#define H_  512
#define L_  5
#define G_  64
#define K1_ 384      // 3*D
#define N2_ 1152     // 2*H + D

// gfx1250 async global->LDS staging. 0 = sync fallback.
#define ASYNC_PATH 1

typedef __attribute__((ext_vector_type(16))) __bf16 v16bf;
typedef __attribute__((ext_vector_type(8)))  float  v8f;

union Frag { v16bf v; uint4 q[2]; };

// LDS B subtile: 64 cols x 32 k, row stride padded to 40 ushorts (80B) so the
// 16-lane ds_load_b128 pattern (col*20 mod 64 banks) is conflict-free.
#define BSTRIDE 40
#define BTILE   (64 * BSTRIDE)

__device__ __forceinline__ unsigned short f32_to_bf16(float f) {
  unsigned int u = __float_as_uint(f);
  u += 0x7FFFu + ((u >> 16) & 1u);   // round-to-nearest-even
  return (unsigned short)(u >> 16);
}

// A fragment straight from global. Per ISA 16-bit layout: lane<16 holds
// K {0..7,16..23}, lane>=16 holds K {8..15,24..31} -> two 16B loads.
__device__ __forceinline__ v16bf load_fragA(const unsigned short* __restrict__ base,
                                            int row, int ld, int kb, int half) {
  const unsigned short* p = base + (size_t)row * ld + kb + half * 8;
  Frag f;
  f.q[0] = *(const uint4*)p;
  f.q[1] = *(const uint4*)(p + 16);
  return f.v;
}

// B fragment from a staged LDS subtile (col = tile-local column).
__device__ __forceinline__ v16bf lds_fragB(const unsigned short* buf, int col, int half) {
  const unsigned short* p = buf + col * BSTRIDE + half * 8;
  Frag f;
  f.q[0] = *(const uint4*)p;
  f.q[1] = *(const uint4*)(p + 16);
  return f.v;
}

// Stage one 64-col x 64-k B macro-tile (8KB) as two 32-k subtiles.
// 256 threads x 2 x 16B. thread t: col=t>>2, seg=t&3.
__device__ __forceinline__ void stage_B2(const unsigned short* __restrict__ Wt,
                                         int n0, int K, int kb,
                                         unsigned short* buf0, unsigned short* buf1,
                                         int tid) {
  int col = tid >> 2;
  int seg = tid & 3;
  const unsigned short* g0 = Wt + (size_t)(n0 + col) * K + kb + seg * 8;
  unsigned short* d0 = buf0 + col * BSTRIDE + seg * 8;
  unsigned short* d1 = buf1 + col * BSTRIDE + seg * 8;
#if ASYNC_PATH
  unsigned o0 = (unsigned)(size_t)d0;   // AS(3) offset lives in low 32 bits
  unsigned o1 = (unsigned)(size_t)d1;
  asm volatile("global_load_async_to_lds_b128 %0, %1, off"
               :: "v"(o0), "v"(g0) : "memory");
  asm volatile("global_load_async_to_lds_b128 %0, %1, off offset:64"
               :: "v"(o1), "v"(g0) : "memory");
#else
  *(uint4*)d0 = *(const uint4*)g0;
  *(uint4*)d1 = *(const uint4*)(g0 + 32);
#endif
}

__device__ __forceinline__ void stage_wait() {
#if ASYNC_PATH
  asm volatile("s_wait_asynccnt 0x0" ::: "memory");
#endif
}

// ---------------- Generic GEMM: C = relu(A @ W + bias) ----------------
// A: MxK bf16, Wt: NxK bf16 (W transposed), out bf16 or f32 (row-major MxN).
// Block = 256 threads = 8 waves; block tile 256(M) x 64(N); wave tile 32x64.
// B staged through double-buffered async LDS (64-k epochs -> 16 WMMA/epoch/wave).
__global__ __launch_bounds__(256)
void k_gemm(const unsigned short* __restrict__ A,
            const unsigned short* __restrict__ Wt,
            const float* __restrict__ bias,
            int M, int K, int N,
            unsigned short* __restrict__ outB,
            float* __restrict__ outF) {
  __shared__ alignas(16) unsigned short sB[2][2][BTILE];
  const int tid  = threadIdx.x;
  const int lane = tid & 31;
  const int wave = tid >> 5;
  const int half = lane >> 4;
  const int l15  = lane & 15;
  const int n0 = blockIdx.x * 64;
  const int m0 = blockIdx.y * 256 + wave * 32;
  int r0 = m0 + l15;      if (r0 >= M) r0 = M - 1;   // clamp loads, guard stores
  int r1 = m0 + 16 + l15; if (r1 >= M) r1 = M - 1;

  stage_B2(Wt, n0, K, 0, sB[0][0], sB[0][1], tid);
  int cur = 0;
  v8f acc0[4] = {}, acc1[4] = {};
  for (int kb = 0; kb < K; kb += 64) {
    stage_wait();
    __syncthreads();
    if (kb + 64 < K) stage_B2(Wt, n0, K, kb + 64, sB[cur ^ 1][0], sB[cur ^ 1][1], tid);
    if (kb + 128 < K) __builtin_prefetch(A + (size_t)r0 * K + kb + 128, 0, 3);
#pragma unroll
    for (int ks = 0; ks < 2; ++ks) {
      v16bf a0 = load_fragA(A, r0, K, kb + ks * 32, half);
      v16bf a1 = load_fragA(A, r1, K, kb + ks * 32, half);
      const unsigned short* sub = sB[cur][ks];
#pragma unroll
      for (int j = 0; j < 4; ++j) {
        v16bf b = lds_fragB(sub, j * 16 + l15, half);
        acc0[j] = __builtin_amdgcn_wmma_f32_16x16x32_bf16(
            false, a0, false, b, (short)0, acc0[j], false, false);
        acc1[j] = __builtin_amdgcn_wmma_f32_16x16x32_bf16(
            false, a1, false, b, (short)0, acc1[j], false, false);
      }
    }
    cur ^= 1;
  }

  const int mh = half * 8;
  float bv[4];
#pragma unroll
  for (int j = 0; j < 4; ++j) bv[j] = bias[n0 + j * 16 + l15];
#pragma unroll
  for (int g = 0; g < 2; ++g) {
    const v8f* acc = g ? acc1 : acc0;
#pragma unroll
    for (int r = 0; r < 8; ++r) {
      int m = m0 + g * 16 + mh + r;
      if (m >= M) continue;
#pragma unroll
      for (int j = 0; j < 4; ++j) {
        int n = n0 + j * 16 + l15;
        float v = acc[j][r] + bv[j];
        v = v > 0.f ? v : 0.f;
        if (outB) outB[(size_t)m * N + n] = f32_to_bf16(v);
        else      outF[(size_t)m * N + n] = v;
      }
    }
  }
}

// ---- GEMM2 with fused scatter epilogue: new_t = relu(h @ w2 + b2); ----
// cols [0,512) -> atomicAdd pooled[s], [512,640) -> pv, [640,1152) -> atomicAdd pooled[o]
__global__ __launch_bounds__(256)
void k_gemm2_scatter(const unsigned short* __restrict__ A,    // T x 512 bf16
                     const unsigned short* __restrict__ Wt,   // 1152 x 512 bf16
                     const float* __restrict__ bias,          // 1152
                     const int* __restrict__ triples,         // T x 3
                     float* __restrict__ pooled,              // O x 512 (pre-zeroed)
                     float* __restrict__ pv) {                // T x 128
  __shared__ alignas(16) unsigned short sB[2][2][BTILE];
  const int M = T_, K = H_;
  const int tid  = threadIdx.x;
  const int lane = tid & 31;
  const int wave = tid >> 5;
  const int half = lane >> 4;
  const int l15  = lane & 15;
  const int n0 = blockIdx.x * 64;
  const int m0 = blockIdx.y * 256 + wave * 32;
  int r0 = m0 + l15;      if (r0 >= M) r0 = M - 1;
  int r1 = m0 + 16 + l15; if (r1 >= M) r1 = M - 1;

  stage_B2(Wt, n0, K, 0, sB[0][0], sB[0][1], tid);
  int cur = 0;
  v8f acc0[4] = {}, acc1[4] = {};
  for (int kb = 0; kb < K; kb += 64) {
    stage_wait();
    __syncthreads();
    if (kb + 64 < K) stage_B2(Wt, n0, K, kb + 64, sB[cur ^ 1][0], sB[cur ^ 1][1], tid);
    if (kb + 128 < K) __builtin_prefetch(A + (size_t)r0 * K + kb + 128, 0, 3);
#pragma unroll
    for (int ks = 0; ks < 2; ++ks) {
      v16bf a0 = load_fragA(A, r0, K, kb + ks * 32, half);
      v16bf a1 = load_fragA(A, r1, K, kb + ks * 32, half);
      const unsigned short* sub = sB[cur][ks];
#pragma unroll
      for (int j = 0; j < 4; ++j) {
        v16bf b = lds_fragB(sub, j * 16 + l15, half);
        acc0[j] = __builtin_amdgcn_wmma_f32_16x16x32_bf16(
            false, a0, false, b, (short)0, acc0[j], false, false);
        acc1[j] = __builtin_amdgcn_wmma_f32_16x16x32_bf16(
            false, a1, false, b, (short)0, acc1[j], false, false);
      }
    }
    cur ^= 1;
  }

  const int mh = half * 8;
  float bv[4];
#pragma unroll
  for (int j = 0; j < 4; ++j) bv[j] = bias[n0 + j * 16 + l15];
#pragma unroll
  for (int g = 0; g < 2; ++g) {
    const v8f* acc = g ? acc1 : acc0;
#pragma unroll
    for (int r = 0; r < 8; ++r) {
      int m = m0 + g * 16 + mh + r;
      if (m >= M) continue;
      int si = triples[3 * m + 0];
      int oi = triples[3 * m + 2];
#pragma unroll
      for (int j = 0; j < 4; ++j) {
        int n = n0 + j * 16 + l15;
        float v = acc[j][r] + bv[j];
        v = v > 0.f ? v : 0.f;
        if (n < H_)            atomicAdd(&pooled[(size_t)si * H_ + n], v);
        else if (n < H_ + D_)  pv[(size_t)m * D_ + (n - H_)] = v;
        else                   atomicAdd(&pooled[(size_t)oi * H_ + (n - H_ - D_)], v);
      }
    }
  }
}

// ---------------- small helper kernels ----------------
__global__ void k_zero(float* p, int n) {
  int i = blockIdx.x * blockDim.x + threadIdx.x;
  if (i < n) p[i] = 0.f;
}

// W (KxN f32) -> Wt (NxK bf16)
__global__ void k_convT(const float* __restrict__ W, unsigned short* __restrict__ Wt,
                        int K, int N) {
  int idx = blockIdx.x * blockDim.x + threadIdx.x;
  if (idx >= K * N) return;
  int n = idx / K, k = idx - n * K;
  Wt[idx] = f32_to_bf16(W[(size_t)k * N + n]);
}

// out[i,:] = emb[idxarr[i*istride+ioff], :]
__global__ void k_gather(const float* __restrict__ emb, const int* __restrict__ idxarr,
                         int istride, int ioff, float* __restrict__ out, int n) {
  int t = blockIdx.x * blockDim.x + threadIdx.x;
  if (t >= n * D_) return;
  int i = t >> 7, d = t & (D_ - 1);
  out[t] = emb[(size_t)idxarr[i * istride + ioff] * D_ + d];
}

// t = bf16(concat(ov[s], pv, ov[o]))
__global__ void k_build_t(const float* __restrict__ ov, const float* __restrict__ pv,
                          const int* __restrict__ triples, unsigned short* __restrict__ t) {
  int idx = blockIdx.x * blockDim.x + threadIdx.x;
  if (idx >= T_ * K1_) return;
  int i = idx / K1_, c = idx - i * K1_;
  float v;
  if (c < D_)            v = ov[(size_t)triples[3 * i + 0] * D_ + c];
  else if (c < 2 * D_)   v = pv[(size_t)i * D_ + (c - D_)];
  else                   v = ov[(size_t)triples[3 * i + 2] * D_ + (c - 2 * D_)];
  t[idx] = f32_to_bf16(v);
}

__global__ void k_counts(const int* __restrict__ triples, float* __restrict__ counts) {
  int i = blockIdx.x * blockDim.x + threadIdx.x;
  if (i >= T_) return;
  atomicAdd(&counts[triples[3 * i + 0]], 1.f);
  atomicAdd(&counts[triples[3 * i + 2]], 1.f);
}

__global__ void k_inv(const float* __restrict__ counts, float* __restrict__ inv) {
  int i = blockIdx.x * blockDim.x + threadIdx.x;
  if (i < O_) inv[i] = 1.f / fmaxf(counts[i], 1.f);
}

// poolb = bf16(pooled * inv[row]); also re-zero pooled for the next layer
__global__ void k_pool_fin(float* __restrict__ pooled, const float* __restrict__ inv,
                           unsigned short* __restrict__ poolb) {
  int idx = blockIdx.x * blockDim.x + threadIdx.x;
  if (idx >= O_ * H_) return;
  int i = idx >> 9;
  poolb[idx] = f32_to_bf16(pooled[idx] * inv[i]);
  pooled[idx] = 0.f;
}

__global__ void k_scores(const float* __restrict__ ov, const float* __restrict__ aw,
                         const float* __restrict__ ab, float* __restrict__ scores) {
  int i = blockIdx.x * blockDim.x + threadIdx.x;
  if (i >= O_) return;
  float s = ab[0];
  for (int d = 0; d < D_; ++d) s += ov[(size_t)i * D_ + d] * aw[d];
  scores[i] = s;
}

// block per segment: gmax[g] = max score in segment; also zero gsum[g]
__global__ void k_segmax(const float* __restrict__ scores, const int* __restrict__ img,
                         float* __restrict__ gmax, float* __restrict__ gsum) {
  __shared__ float red[256];
  int g = blockIdx.x;
  float m = -3.0e38f;
  for (int i = threadIdx.x; i < O_; i += blockDim.x)
    if (img[i] == g) m = fmaxf(m, scores[i]);
  red[threadIdx.x] = m;
  __syncthreads();
  for (int s = 128; s > 0; s >>= 1) {
    if ((int)threadIdx.x < s) red[threadIdx.x] = fmaxf(red[threadIdx.x], red[threadIdx.x + s]);
    __syncthreads();
  }
  if (threadIdx.x == 0) { gmax[g] = red[0]; gsum[g] = 0.f; }
}

// scores[i] <- e_i = exp(score - gmax); gsum[g] += e_i
__global__ void k_expsum(float* __restrict__ scores, const int* __restrict__ img,
                         const float* __restrict__ gmax, float* __restrict__ gsum) {
  int i = blockIdx.x * blockDim.x + threadIdx.x;
  if (i >= O_) return;
  int g = img[i];
  float e = expf(scores[i] - gmax[g]);
  scores[i] = e;
  atomicAdd(&gsum[g], e);
}

// gg[g,d] += (e_i/gsum[g]) * ov[i,d]
__global__ void k_gg(const float* __restrict__ e, const float* __restrict__ gsum,
                     const int* __restrict__ img, const float* __restrict__ ov,
                     float* __restrict__ gg) {
  int idx = blockIdx.x * blockDim.x + threadIdx.x;
  if (idx >= O_ * D_) return;
  int i = idx >> 7, d = idx & (D_ - 1);
  int g = img[i];
  atomicAdd(&gg[(size_t)g * D_ + d], (e[i] / gsum[g]) * ov[idx]);
}

__global__ void k_final(const float* __restrict__ ov, const float* __restrict__ gg,
                        const int* __restrict__ img, float* __restrict__ out) {
  int idx = blockIdx.x * blockDim.x + threadIdx.x;
  if (idx >= O_ * 2 * D_) return;
  int i = idx >> 8, c = idx & (2 * D_ - 1);
  out[idx] = (c < D_) ? ov[(size_t)i * D_ + c]
                      : gg[(size_t)img[i] * D_ + (c - D_)];
}

// ---------------- launcher ----------------
static inline dim3 g1(int n) { return dim3((n + 255) / 256); }

extern "C" void kernel_launch(void* const* d_in, const int* in_sizes, int n_in,
                              void* d_out, int out_size, void* d_ws, size_t ws_size,
                              hipStream_t stream) {
  const int*   objs     = (const int*)d_in[0];
  const int*   triples  = (const int*)d_in[1];
  const int*   img      = (const int*)d_in[2];
  const float* obj_emb  = (const float*)d_in[3];
  const float* pred_emb = (const float*)d_in[4];
  const float* n1w1     = (const float*)d_in[5];
  const float* n1b1     = (const float*)d_in[6];
  const float* n1w2     = (const float*)d_in[7];
  const float* n1b2     = (const float*)d_in[8];
  const float* n2w1     = (const float*)d_in[9];
  const float* n2b1     = (const float*)d_in[10];
  const float* n2w2     = (const float*)d_in[11];
  const float* n2b2     = (const float*)d_in[12];
  const float* att_w    = (const float*)d_in[13];
  const float* att_b    = (const float*)d_in[14];

  // workspace carve-up (~72 MB)
  char* ws = (char*)d_ws;
  size_t off = 0;
  auto alloc = [&](size_t bytes) -> char* {
    char* p = ws + off; off += (bytes + 255) & ~(size_t)255; return p;
  };
  float*          ov     = (float*)alloc((size_t)O_ * D_ * 4);
  float*          pv     = (float*)alloc((size_t)T_ * D_ * 4);
  unsigned short* tb     = (unsigned short*)alloc((size_t)T_ * K1_ * 2);
  unsigned short* hb     = (unsigned short*)alloc((size_t)T_ * H_ * 2);
  float*          pooled = (float*)alloc((size_t)O_ * H_ * 4);
  unsigned short* poolb  = (unsigned short*)alloc((size_t)O_ * H_ * 2);
  unsigned short* h2b    = (unsigned short*)alloc((size_t)O_ * H_ * 2);
  unsigned short* w1t    = (unsigned short*)alloc((size_t)H_ * K1_ * 2);
  unsigned short* w2t    = (unsigned short*)alloc((size_t)N2_ * H_ * 2);
  unsigned short* w3t    = (unsigned short*)alloc((size_t)H_ * H_ * 2);
  unsigned short* w4t    = (unsigned short*)alloc((size_t)D_ * H_ * 2);
  float*          counts = (float*)alloc((size_t)O_ * 4);
  float*          inv    = (float*)alloc((size_t)O_ * 4);
  float*          scores = (float*)alloc((size_t)O_ * 4);
  float*          gmax   = (float*)alloc((size_t)G_ * 4);
  float*          gsum   = (float*)alloc((size_t)G_ * 4);
  float*          gg     = (float*)alloc((size_t)G_ * D_ * 4);

  // initial gathers + counts (layer-invariant)
  k_gather<<<g1(O_ * D_), 256, 0, stream>>>(obj_emb, objs, 1, 0, ov, O_);
  k_gather<<<g1(T_ * D_), 256, 0, stream>>>(pred_emb, triples, 3, 1, pv, T_);
  k_zero<<<g1(O_), 256, 0, stream>>>(counts, O_);
  k_zero<<<g1(O_ * H_), 256, 0, stream>>>(pooled, O_ * H_);
  k_counts<<<g1(T_), 256, 0, stream>>>(triples, counts);
  k_inv<<<g1(O_), 256, 0, stream>>>(counts, inv);

  const dim3 B(256);
  for (int l = 0; l < L_; ++l) {
    k_convT<<<g1(K1_ * H_), 256, 0, stream>>>(n1w1 + (size_t)l * K1_ * H_, w1t, K1_, H_);
    k_convT<<<g1(H_ * N2_), 256, 0, stream>>>(n1w2 + (size_t)l * H_ * N2_, w2t, H_, N2_);
    k_convT<<<g1(H_ * H_),  256, 0, stream>>>(n2w1 + (size_t)l * H_ * H_,  w3t, H_, H_);
    k_convT<<<g1(H_ * D_),  256, 0, stream>>>(n2w2 + (size_t)l * H_ * D_,  w4t, H_, D_);

    k_build_t<<<g1(T_ * K1_), 256, 0, stream>>>(ov, pv, triples, tb);

    // MLP1a: h = relu(t @ w1 + b1), bf16 out
    k_gemm<<<dim3(H_ / 64, (T_ + 255) / 256), B, 0, stream>>>(
        tb, w1t, n1b1 + (size_t)l * H_, T_, K1_, H_, hb, nullptr);
    // MLP1b fused with scatter-pool
    k_gemm2_scatter<<<dim3(N2_ / 64, (T_ + 255) / 256), B, 0, stream>>>(
        hb, w2t, n1b2 + (size_t)l * N2_, triples, pooled, pv);
    // normalize + convert, re-zero pooled
    k_pool_fin<<<g1(O_ * H_), 256, 0, stream>>>(pooled, inv, poolb);
    // MLP2a: h2 = relu(pooled @ w3 + b3), bf16 out
    k_gemm<<<dim3(H_ / 64, (O_ + 255) / 256), B, 0, stream>>>(
        poolb, w3t, n2b1 + (size_t)l * H_, O_, H_, H_, h2b, nullptr);
    // MLP2b: ov = relu(h2 @ w4 + b4), f32 out
    k_gemm<<<dim3(D_ / 64, (O_ + 255) / 256), B, 0, stream>>>(
        h2b, w4t, n2b2 + (size_t)l * D_, O_, H_, D_, nullptr, ov);
  }

  // segment-softmax attention pooling
  k_scores<<<g1(O_), 256, 0, stream>>>(ov, att_w, att_b, scores);
  k_segmax<<<G_, 256, 0, stream>>>(scores, img, gmax, gsum);
  k_zero<<<g1(G_ * D_), 256, 0, stream>>>(gg, G_ * D_);
  k_expsum<<<g1(O_), 256, 0, stream>>>(scores, img, gmax, gsum);
  k_gg<<<g1(O_ * D_), 256, 0, stream>>>(scores, gsum, img, ov, gg);
  k_final<<<g1(O_ * 2 * D_), 256, 0, stream>>>(ov, gg, img, (float*)d_out);
}
